// GAT_63934883168891
// MI455X (gfx1250) — compile-verified, matrix operand order checked by reading, not run on hardware
//
#include <hip/hip_runtime.h>
#include <hip/hip_bf16.h>
#include <math.h>

#define L_LAYERS 4
#define H_HEADS  4
#define D_DIM    32
#define IN_F     128
#define HD       128
#define NEG_SLOPE 0.2f

typedef __attribute__((ext_vector_type(16))) __bf16 v16bf;
typedef __attribute__((ext_vector_type(8)))  float  v8f;

__device__ __forceinline__ unsigned short f2bf(float f) {
  // round-to-nearest-even f32 -> bf16
  unsigned int x = __float_as_uint(f);
  unsigned int r = x + 0x7FFFu + ((x >> 16) & 1u);
  return (unsigned short)(r >> 16);
}

__device__ __forceinline__ void atomicMaxF(float* addr, float value) {
  // works with -inf init: nonneg values order as signed ints, negatives order
  // inversely as unsigned ints.
  if (value >= 0.0f)
    atomicMax((int*)addr, __float_as_int(value));
  else
    atomicMin((unsigned int*)addr, __float_as_uint(value));
}

// ---------------------------------------------------------------------------
// Ws (L x IN x HD, row-major f32) -> WsT (L x HD x IN, bf16 col-major)
__global__ void k_convert_ws(const float* __restrict__ Ws,
                             unsigned short* __restrict__ WsT, int total) {
  int i = blockIdx.x * blockDim.x + threadIdx.x;
  if (i >= total) return;                    // total = L*IN*HD
  int l = i / (IN_F * HD);
  int r = i % (IN_F * HD);
  int k = r / HD;                            // input-feature row
  int n = r % HD;                            // output col
  WsT[(size_t)l * IN_F * HD + (size_t)n * IN_F + k] = f2bf(Ws[i]);
}

// h[n][c] = embed[x[n]][c]; also produce bf16 copy for the WMMA GEMM
__global__ void k_gather(const int* __restrict__ x, const float* __restrict__ embed,
                         float* __restrict__ h, unsigned short* __restrict__ hb,
                         int N) {
  int i = blockIdx.x * blockDim.x + threadIdx.x;
  if (i >= N * IN_F) return;
  int n = i >> 7, c = i & 127;
  float v = embed[(size_t)x[n] * IN_F + c];
  h[i]  = v;
  hb[i] = f2bf(v);
}

// ---------------------------------------------------------------------------
// feat = h @ Ws[l]  via v_wmma_f32_16x16x32_bf16.
// One wave per 16x16 C tile; 8 waves/block cover one 16-row panel x 128 cols.
// A and B are pre-converted bf16 with K contiguous per lane, so each fragment
// half loads as a single global_load_b128.
__global__ __launch_bounds__(256) void k_gemm_wmma(
    const unsigned short* __restrict__ hb, const unsigned short* __restrict__ WsT,
    float* __restrict__ feat, int N) {
  const int lane  = threadIdx.x & 31;
  const int wave  = threadIdx.x >> 5;        // 0..7 -> N-tile
  const int mtile = blockIdx.x;
  const int half  = lane >> 4;
  const int l15   = lane & 15;
  const int arow  = mtile * 16 + l15;        // A row this lane supplies
  const int arowc = arow < N ? arow : N - 1; // clamp (only pollutes unstored rows)
  const int bcol  = wave * 16 + l15;         // C column this lane owns

  const uint4* Arow = (const uint4*)(hb  + (size_t)arowc * IN_F); // 8 bf16 per uint4
  const uint4* Bcol = (const uint4*)(WsT + (size_t)bcol  * IN_F);

  v8f acc = {};
  #pragma unroll
  for (int kc = 0; kc < IN_F; kc += 32) {
    // ISA 16-bit A 16x32 layout (lanes 0-15 / 16-31 split by `half`):
    //   regs 0-3: K = kc +    8*half + {0..7}   -> one b128
    //   regs 4-7: K = kc + 16 + 8*half + {0..7} -> one b128
    union { v16bf v; uint4 q[2]; } A, B;
    A.q[0] = Arow[(kc + 8 * half) >> 3];
    A.q[1] = Arow[(kc + 16 + 8 * half) >> 3];
    B.q[0] = Bcol[(kc + 8 * half) >> 3];
    B.q[1] = Bcol[(kc + 16 + 8 * half) >> 3];
    acc = __builtin_amdgcn_wmma_f32_16x16x32_bf16(
        /*neg_a=*/false, A.v, /*neg_b=*/false, B.v,
        /*c_mod=*/(short)0, acc, /*reuse_a=*/false, /*reuse_b=*/false);
  }
  // C/D layout: VGPR r -> M = r + 8*half, N = lane&15.
  // Base row for this lane's stores; stores are at rows m0+r, column bcol.
  const int m0 = mtile * 16 + half * 8;
  float* outp = feat + (size_t)m0 * HD + bcol;
  if (mtile * 16 + 15 < N) {
    // full tile (always taken when N % 16 == 0): unguarded store clause
    #pragma unroll
    for (int r = 0; r < 8; ++r) outp[(size_t)r * HD] = acc[r];
  } else {
    #pragma unroll
    for (int r = 0; r < 8; ++r)
      if (m0 + r < N) outp[(size_t)r * HD] = acc[r];
  }
}

// el/er[n][h] = <feat[n][h][:], attn_l/r[h][:]>
__global__ void k_attn(const float* __restrict__ feat, const float* __restrict__ al,
                       const float* __restrict__ ar, float* __restrict__ el,
                       float* __restrict__ er, int N) {
  int i = blockIdx.x * blockDim.x + threadIdx.x;   // over N*H
  if (i >= N * H_HEADS) return;
  int n = i >> 2, hh = i & 3;
  const float* f  = feat + (size_t)n * HD + hh * D_DIM;
  const float* wl = al + hh * D_DIM;
  const float* wr = ar + hh * D_DIM;
  float sl = 0.f, sr = 0.f;
  #pragma unroll
  for (int d = 0; d < D_DIM; ++d) { float v = f[d]; sl += v * wl[d]; sr += v * wr[d]; }
  el[i] = sl;
  er[i] = sr;
}

__global__ void k_init_layer(float* __restrict__ m, float* __restrict__ s,
                             float* __restrict__ rst, int N) {
  int i = blockIdx.x * blockDim.x + threadIdx.x;   // over N*HD
  if (i >= N * HD) return;
  rst[i] = 0.f;
  if (i < N * H_HEADS) { m[i] = -__builtin_inff(); s[i] = 0.f; }
}

__global__ void k_edge_max(const int* __restrict__ src, const int* __restrict__ dst,
                           const float* __restrict__ el, const float* __restrict__ er,
                           float* __restrict__ m, int E) {
  int i = blockIdx.x * blockDim.x + threadIdx.x;   // over E*H
  if (i >= E * H_HEADS) return;
  int e = i >> 2, hh = i & 3;
  int se = src[e], de = dst[e];
  float v = el[se * H_HEADS + hh] + er[de * H_HEADS + hh];
  v = v > 0.f ? v : NEG_SLOPE * v;
  atomicMaxF(&m[de * H_HEADS + hh], v);
}

__global__ void k_edge_sum(const int* __restrict__ src, const int* __restrict__ dst,
                           const float* __restrict__ el, const float* __restrict__ er,
                           const float* __restrict__ m, float* __restrict__ s, int E) {
  int i = blockIdx.x * blockDim.x + threadIdx.x;   // over E*H
  if (i >= E * H_HEADS) return;
  int e = i >> 2, hh = i & 3;
  int se = src[e], de = dst[e];
  float v = el[se * H_HEADS + hh] + er[de * H_HEADS + hh];
  v = v > 0.f ? v : NEG_SLOPE * v;
  atomicAdd(&s[de * H_HEADS + hh], __expf(v - m[de * H_HEADS + hh]));
}

__global__ void k_edge_msg(const int* __restrict__ src, const int* __restrict__ dst,
                           const float* __restrict__ el, const float* __restrict__ er,
                           const float* __restrict__ m, const float* __restrict__ s,
                           const float* __restrict__ feat, float* __restrict__ rst,
                           int E) {
  int i = blockIdx.x * blockDim.x + threadIdx.x;   // over E*HD (=102.4M < 2^31)
  if (i >= E * HD) return;
  int e = i >> 7, c = i & 127, hh = c >> 5;
  int se = src[e], de = dst[e];
  float v = el[se * H_HEADS + hh] + er[de * H_HEADS + hh];
  v = v > 0.f ? v : NEG_SLOPE * v;
  float alpha = __expf(v - m[de * H_HEADS + hh]) / s[de * H_HEADS + hh];
  atomicAdd(&rst[(size_t)de * HD + c], feat[(size_t)se * HD + c] * alpha);
}

// h = relu(rst); also refresh the bf16 copy for the next layer's GEMM
__global__ void k_relu(const float* __restrict__ rst, float* __restrict__ h,
                       unsigned short* __restrict__ hb, int total) {
  int i = blockIdx.x * blockDim.x + threadIdx.x;
  if (i >= total) return;
  float v = fmaxf(rst[i], 0.f);
  h[i]  = v;
  hb[i] = f2bf(v);
}

// ---------------------------------------------------------------------------
__global__ void k_init_final(float* gmax, float* gsum, float* tmp) {
  int i = threadIdx.x;
  if (i == 0) { *gmax = -__builtin_inff(); *gsum = 0.f; }
  if (i < HD) tmp[i] = 0.f;
}

__global__ void k_gate(const float* __restrict__ h, const float* __restrict__ gw,
                       const float* __restrict__ gb, float* __restrict__ gate,
                       float* __restrict__ gmax, int N) {
  int n = blockIdx.x * blockDim.x + threadIdx.x;
  if (n >= N) return;
  float acc = gb[0];
  #pragma unroll 4
  for (int c = 0; c < HD; ++c) acc += h[(size_t)n * HD + c] * gw[c];
  float g = 1.f / (1.f + __expf(-acc));
  gate[n] = g;
  atomicMaxF(gmax, g);
}

__global__ void k_pexp(const float* __restrict__ gate, const float* __restrict__ gmax,
                       float* __restrict__ p, float* __restrict__ gsum, int N) {
  int n = blockIdx.x * blockDim.x + threadIdx.x;
  if (n >= N) return;
  float v = __expf(gate[n] - *gmax);
  p[n] = v;
  atomicAdd(gsum, v);
}

__global__ void k_accum(const float* __restrict__ h, const float* __restrict__ p,
                        float* __restrict__ tmp, int N) {
  int c = threadIdx.x;                       // 128 threads = channels
  float acc = 0.f;
  for (int n = blockIdx.x; n < N; n += gridDim.x)
    acc += p[n] * h[(size_t)n * HD + c];
  atomicAdd(&tmp[c], acc);
}

__global__ void k_final(const float* __restrict__ tmp, const float* __restrict__ gsum,
                        float* __restrict__ out) {
  int c = threadIdx.x;
  if (c < HD) out[c] = tmp[c] / *gsum;
}

// ---------------------------------------------------------------------------
extern "C" void kernel_launch(void* const* d_in, const int* in_sizes, int n_in,
                              void* d_out, int out_size, void* d_ws, size_t ws_size,
                              hipStream_t stream) {
  const int*   x     = (const int*)  d_in[0];
  const int*   src   = (const int*)  d_in[1];
  const int*   dst   = (const int*)  d_in[2];
  const float* embed = (const float*)d_in[3];
  const float* Ws    = (const float*)d_in[4];
  const float* al    = (const float*)d_in[5];
  const float* ar    = (const float*)d_in[6];
  const float* gw    = (const float*)d_in[7];
  const float* gb    = (const float*)d_in[8];
  float* out = (float*)d_out;

  const int N = in_sizes[0];
  const int E = in_sizes[1];

  // workspace carve-up
  char* ws = (char*)d_ws;
  size_t off = 0;
  auto carve = [&](size_t bytes) -> char* {
    char* p = ws + off;
    off = (off + bytes + 255) & ~(size_t)255;
    return p;
  };
  float*          h_buf  = (float*)carve((size_t)N * HD * sizeof(float));
  unsigned short* hb     = (unsigned short*)carve((size_t)N * HD * sizeof(unsigned short));
  float*          feat   = (float*)carve((size_t)N * HD * sizeof(float));
  float*          rst    = (float*)carve((size_t)N * HD * sizeof(float));
  float*          el     = (float*)carve((size_t)N * H_HEADS * sizeof(float));
  float*          er     = (float*)carve((size_t)N * H_HEADS * sizeof(float));
  float*          mbuf   = (float*)carve((size_t)N * H_HEADS * sizeof(float));
  float*          sbuf   = (float*)carve((size_t)N * H_HEADS * sizeof(float));
  unsigned short* WsT    = (unsigned short*)carve((size_t)L_LAYERS * IN_F * HD * sizeof(unsigned short));
  float*          gate   = (float*)carve((size_t)N * sizeof(float));
  float*          pbuf   = (float*)carve((size_t)N * sizeof(float));
  float*          tmp    = (float*)carve(HD * sizeof(float));
  float*          gmax   = (float*)carve(sizeof(float));
  float*          gsum   = (float*)carve(sizeof(float));

  const int B = 256;
  auto blks = [&](long long n) { return (int)((n + B - 1) / B); };

  k_convert_ws<<<blks((long long)L_LAYERS * IN_F * HD), B, 0, stream>>>(
      Ws, WsT, L_LAYERS * IN_F * HD);
  k_gather<<<blks((long long)N * IN_F), B, 0, stream>>>(x, embed, h_buf, hb, N);

  const int mtiles = (N + 15) / 16;
  for (int l = 0; l < L_LAYERS; ++l) {
    k_init_layer<<<blks((long long)N * HD), B, 0, stream>>>(mbuf, sbuf, rst, N);
    k_gemm_wmma<<<mtiles, 256, 0, stream>>>(
        hb, WsT + (size_t)l * IN_F * HD, feat, N);
    k_attn<<<blks((long long)N * H_HEADS), B, 0, stream>>>(
        feat, al + (size_t)l * H_HEADS * D_DIM, ar + (size_t)l * H_HEADS * D_DIM,
        el, er, N);
    k_edge_max<<<blks((long long)E * H_HEADS), B, 0, stream>>>(src, dst, el, er, mbuf, E);
    k_edge_sum<<<blks((long long)E * H_HEADS), B, 0, stream>>>(src, dst, el, er, mbuf, sbuf, E);
    k_edge_msg<<<blks((long long)E * HD), B, 0, stream>>>(
        src, dst, el, er, mbuf, sbuf, feat, rst, E);
    k_relu<<<blks((long long)N * HD), B, 0, stream>>>(rst, h_buf, hb, N * HD);
  }

  k_init_final<<<1, 128, 0, stream>>>(gmax, gsum, tmp);
  k_gate<<<blks(N), B, 0, stream>>>(h_buf, gw, gb, gate, gmax, N);
  k_pexp<<<blks(N), B, 0, stream>>>(gate, gmax, pbuf, gsum, N);
  k_accum<<<256, 128, 0, stream>>>(h_buf, pbuf, tmp, N);
  k_final<<<1, 128, 0, stream>>>(tmp, gsum, out);
}
